// CrossAttention_53395033424536
// MI455X (gfx1250) — matrix operand order repaired and checked by reading, not compiled, not run on hardware
//
#include <hip/hip_runtime.h>

#define DEVINL __device__ __forceinline__

typedef __attribute__((ext_vector_type(16))) __bf16 v16bf;
typedef __attribute__((ext_vector_type(8)))  __bf16 v8bf;
typedef __attribute__((ext_vector_type(8)))  float  v8f;

typedef __attribute__((address_space(3))) __bf16       lds_bf16;
typedef __attribute__((address_space(3))) const v8bf   lds_cv8bf;
typedef __attribute__((address_space(3))) const uint4  lds_cu4;

// ---------- helpers ----------

DEVINL __bf16 f2bf(float f) {
  union { float f; unsigned u; } x; x.f = f;
  unsigned r = x.u + 0x7FFFu + ((x.u >> 16) & 1u);   // round-to-nearest-even
  unsigned short h = (unsigned short)(r >> 16);
  return __builtin_bit_cast(__bf16, h);
}

DEVINL unsigned lds_base(const void* p) {
  return (unsigned)(unsigned long long)p;  // generic LDS addr: low 32 bits = LDS offset
}
// Forced-DS accessors: build addrspace(3) pointers from 32-bit LDS byte offsets so
// the backend always emits ds_load/ds_store (never flat_*), keeping WMMA feeds on
// the DScnt-only path.
DEVINL v8bf lds_load8(unsigned byteoff) { return *(lds_cv8bf*)byteoff; }
DEVINL void lds_store2(unsigned byteoff, __bf16 v) { *(lds_bf16*)byteoff = v; }

// A-fragment (16x32 bf16): lane holds row m=(lane&15); lanes 0-15: K = k0+{0..7,16..23},
// lanes 16-31: K = k0+{8..15,24..31}.  Two 16B DS loads.
DEVINL v16bf frag_a(unsigned tile, int m0, int k0, int ldk) {
  const int lane = threadIdx.x & 31;
  const unsigned row =
      tile + (unsigned)((m0 + (lane & 15)) * ldk + k0 + ((lane >> 4) << 3)) * 2u;
  v8bf lo = lds_load8(row);
  v8bf hi = lds_load8(row + 32);
  v16bf a;
#pragma unroll
  for (int i = 0; i < 8; ++i) { a[i] = lo[i]; a[i + 8] = hi[i]; }
  return a;
}

// B-fragment (32x16 bf16) from LDS stored [n][k]: lane holds col n=(lane&15);
// lanes 0-15: K = k0+{0..7,16..23}, lanes 16-31: K = k0+{8..15,24..31}.  Two 16B DS loads.
DEVINL v16bf frag_b(unsigned tile, int n0, int k0, int ldk) {
  const int lane = threadIdx.x & 31;
  const unsigned row =
      tile + (unsigned)((n0 + (lane & 15)) * ldk + k0 + ((lane >> 4) << 3)) * 2u;
  v8bf lo = lds_load8(row);
  v8bf hi = lds_load8(row + 32);
  v16bf b;
#pragma unroll
  for (int i = 0; i < 8; ++i) { b[i] = lo[i]; b[i + 8] = hi[i]; }
  return b;
}

DEVINL v8f wmma_bf16(v16bf a, v16bf b, v8f c) {
  return __builtin_amdgcn_wmma_f32_16x16x32_bf16(false, a, false, b, (short)0, c,
                                                 false, false);
}

// gfx1250 async global -> LDS (GVS mode: vdst = LDS byte offset, vaddr = 32b byte
// offset, saddr = uniform 64b base).  Tracked by ASYNCcnt.
DEVINL void async_ld128(unsigned ldsoff, unsigned goff, const void* sbase) {
  asm volatile("global_load_async_to_lds_b128 %0, %1, %2"
               :: "v"(ldsoff), "v"(goff), "s"(sbase) : "memory");
}
DEVINL void wait_async0() { asm volatile("s_wait_asynccnt 0x0" ::: "memory"); }

// ---------- generic projection GEMM (register + LDS double-buffered) ----------
// C[M,N] = A[M,K] * W[K,N] + bias, block tile 128x128, K-step 32.
// EPI 0: Q-proj -> bias, RoPE, *scale, bf16 scatter [B,H,seq,128]
// EPI 1: K/V-proj -> bias, bf16 scatter [B,H,seq,128]
// EPI 2: O-proj  -> bias, fp32 row-major out
template <int EPI, bool ABF16>
__global__ __launch_bounds__(256, 1) void gemm_wmma(
    const float* __restrict__ Af, const __bf16* __restrict__ Ab,
    const float* __restrict__ W, const float* __restrict__ bias,
    void* __restrict__ outp, int M, int K, int N, int seq, float scale) {
  extern __shared__ __align__(16) char gsm[];
  (void)M;
  // LDS byte-offset map: A buffers [2][128*40], then B buffers [2][128*40]
  const unsigned SM   = lds_base(gsm);
  const unsigned ABUF = 128 * 40 * 2;          // 10240 B per buffer
  const unsigned BOFF = SM + 2 * ABUF;

  const int tid  = threadIdx.x;
  const int lane = tid & 31;
  const int w    = tid >> 5;
  const int wm   = w >> 2;           // 0..1 -> 64 rows
  const int wn   = w & 3;            // 0..3 -> 32 cols
  const int m0   = blockIdx.y * 128;
  const int n0   = blockIdx.x * 128;

  v8f acc[4][2];
#pragma unroll
  for (int mi = 0; mi < 4; ++mi)
#pragma unroll
    for (int nj = 0; nj < 2; ++nj)
#pragma unroll
      for (int r = 0; r < 8; ++r) acc[mi][nj][r] = 0.0f;

  __bf16 ar[16], br[16];             // staging registers (double-buffer front end)
  auto load_tiles = [&](int k0) {
#pragma unroll
    for (int i = 0; i < 16; ++i) {
      int e = tid + 256 * i;
      int m = e >> 5, k = e & 31;
      if constexpr (ABF16) ar[i] = Ab[(size_t)(m0 + m) * K + k0 + k];
      else                 ar[i] = f2bf(Af[(size_t)(m0 + m) * K + k0 + k]);
    }
#pragma unroll
    for (int i = 0; i < 16; ++i) {
      int e = tid + 256 * i;
      int k = e >> 7, n = e & 127;
      br[i] = f2bf(W[(size_t)(k0 + k) * N + n0 + n]);
    }
    if (k0 + 32 < K)  // warm L2 for the next weight panel (global_prefetch_b8)
      __builtin_prefetch(W + (size_t)(k0 + 32) * N + n0 + ((tid & 31) << 2), 0, 0);
  };

  load_tiles(0);
  for (int k0 = 0; k0 < K; k0 += 32) {
    const unsigned buf = ((unsigned)k0 >> 5) & 1u;
    const unsigned Ac  = SM + buf * ABUF;
    const unsigned Bc  = BOFF + buf * ABUF;
#pragma unroll
    for (int i = 0; i < 16; ++i) {
      int e = tid + 256 * i;
      lds_store2(Ac + (unsigned)((e >> 5) * 40 + (e & 31)) * 2u, ar[i]);
      lds_store2(Bc + (unsigned)((e & 127) * 40 + (e >> 7)) * 2u, br[i]);
    }
    __syncthreads();
    if (k0 + 32 < K) load_tiles(k0 + 32);   // overlap next global loads with WMMAs

    v16bf bfr[2];
#pragma unroll
    for (int nj = 0; nj < 2; ++nj) bfr[nj] = frag_b(Bc, wn * 32 + nj * 16, 0, 40);
#pragma unroll
    for (int mi = 0; mi < 4; ++mi) {
      v16bf af = frag_a(Ac, wm * 64 + mi * 16, 0, 40);
#pragma unroll
      for (int nj = 0; nj < 2; ++nj) acc[mi][nj] = wmma_bf16(af, bfr[nj], acc[mi][nj]);
    }
  }

  // epilogue
#pragma unroll
  for (int mi = 0; mi < 4; ++mi) {
#pragma unroll
    for (int nj = 0; nj < 2; ++nj) {
      const int gn    = n0 + wn * 32 + nj * 16 + (lane & 15);
      const float bvv = bias[gn];
      const int mbase = m0 + wm * 64 + mi * 16 + ((lane >> 4) << 3);
#pragma unroll
      for (int r = 0; r < 8; ++r) {
        const int gm = mbase + r;
        float v = acc[mi][nj][r] + bvv;
        if constexpr (EPI == 2) {
          ((float*)outp)[(size_t)gm * N + gn] = v;
        } else {
          const int bidx = gm / seq, s = gm % seq;
          const int h = gn >> 7, hd = gn & 127;
          if constexpr (EPI == 0) {
            float vo = __shfl_xor(v, 1);  // partner of the (even,odd) rope pair
            float fi  = (float)(hd & ~1);
            float ang = (float)s * __powf(10000.0f, -fi * (1.0f / 128.0f));
            float sn, cs; __sincosf(ang, &sn, &cs);
            float rv = (hd & 1) ? (vo * sn + v * cs) : (v * cs - vo * sn);
            v = rv * scale;               // fold 1/sqrt(HD) into Q
          }
          ((__bf16*)outp)[((size_t)(bidx * 16 + h) * seq + s) * 128 + hd] = f2bf(v);
        }
      }
    }
  }
}

// ---------- fused flash attention: one block per (b,h) ----------
// Q[256,128] resident in LDS (fragments hoisted to VGPRs).  K/V tiles (64 keys)
// double-buffered and streamed with GLOBAL_LOAD_ASYNC_TO_LDS_B128 overlapping
// the WMMA + online-softmax compute; retired with s_wait_asynccnt.
__global__ __launch_bounds__(256, 1) void attn_fused(
    const __bf16* __restrict__ Qg, const __bf16* __restrict__ Kg,
    const __bf16* __restrict__ Vg, __bf16* __restrict__ ctxo) {
  const int bh = blockIdx.x, b = bh >> 4, h = bh & 15;
  const int tid = threadIdx.x, lane = tid & 31, w = tid >> 5;

  extern __shared__ __align__(16) char smem[];
  // LDS byte-offset map (no pointer arrays -> stays on the DS path):
  const unsigned SM    = lds_base(smem);
  const unsigned QS    = SM;                       // [256][128]            65536 B
  const unsigned KS0   = SM + 65536u;              // [64][128] x2          16384 B each
  const unsigned VR0   = SM + 98304u;              // [64][128] x2 raw V
  const unsigned VTS   = SM + 131072u;             // [128][72] transposed V 18432 B
  const unsigned PS    = SM + 149504u + (unsigned)w * 4096u;  // per-wave [32][64]

  const __bf16* Qb = Qg + (size_t)bh * 256 * 128;
  const __bf16* Kb = Kg + (size_t)bh * 2048 * 128;
  const __bf16* Vb = Vg + (size_t)bh * 2048 * 128;

  // async-stage Q (64KB) + first K/V tile, then one wait + barrier
#pragma unroll
  for (int i = 0; i < 16; ++i) {
    const unsigned o = (unsigned)(tid + 256 * i) * 16u;
    async_ld128(QS + o, o, Qb);
  }
#pragma unroll
  for (int i = 0; i < 4; ++i) {
    const unsigned o = (unsigned)(tid + 256 * i) * 16u;
    async_ld128(KS0 + o, o, Kb);
    async_ld128(VR0 + o, o, Vb);
  }
  wait_async0();
  __syncthreads();

  // hoist Q fragments: wave owns rows 32w..32w+31 (2 M-tiles), 4 K-steps
  v16bf qf[2][4];
#pragma unroll
  for (int t = 0; t < 2; ++t)
#pragma unroll
    for (int ks = 0; ks < 4; ++ks) qf[t][ks] = frag_a(QS, w * 32 + t * 16, ks * 32, 128);

  v8f   o[2][8];
  float mrun[2][8], lrun[2][8];
#pragma unroll
  for (int t = 0; t < 2; ++t)
#pragma unroll
    for (int r = 0; r < 8; ++r) {
      mrun[t][r] = -3.0e38f; lrun[t][r] = 0.0f;
#pragma unroll
      for (int nj = 0; nj < 8; ++nj) o[t][nj][r] = 0.0f;
    }

  for (int it = 0; it < 32; ++it) {
    const unsigned cur = (unsigned)it & 1u, nxt = cur ^ 1u;
    const unsigned KSc = KS0 + cur * 16384u;
    const unsigned VRc = VR0 + cur * 16384u;

    // transpose Vraw[cur] -> VTs[hd][key] (DS->DS; prior readers fenced by the
    // barrier at the end of the previous iteration)
#pragma unroll
    for (int i = 0; i < 4; ++i) {
      int idx = tid + 256 * i;
      int key = idx >> 4, hd0 = (idx & 15) << 3;
      v8bf vv = lds_load8(VRc + (unsigned)(key * 128 + hd0) * 2u);
#pragma unroll
      for (int j = 0; j < 8; ++j)
        lds_store2(VTS + (unsigned)((hd0 + j) * 72 + key) * 2u, vv[j]);
    }
    if (it + 1 < 32) {  // stream next K/V tile while this one is consumed
      const __bf16* kg = Kb + (size_t)(it + 1) * 64 * 128;  // uniform -> SGPR base
      const __bf16* vg = Vb + (size_t)(it + 1) * 64 * 128;
      const unsigned kofs = KS0 + nxt * 16384u;
      const unsigned vofs = VR0 + nxt * 16384u;
#pragma unroll
      for (int i = 0; i < 4; ++i) {
        const unsigned oo = (unsigned)(tid + 256 * i) * 16u;
        async_ld128(kofs + oo, oo, kg);
        async_ld128(vofs + oo, oo, vg);
      }
    }
    __syncthreads();

    // S = Q * K^T  (scale pre-folded into Q)
    v8f s[2][4];
#pragma unroll
    for (int t = 0; t < 2; ++t)
#pragma unroll
      for (int j = 0; j < 4; ++j)
#pragma unroll
        for (int r = 0; r < 8; ++r) s[t][j][r] = 0.0f;
#pragma unroll
    for (int ks = 0; ks < 4; ++ks)
#pragma unroll
      for (int j = 0; j < 4; ++j) {
        v16bf bk = frag_b(KSc, j * 16, ks * 32, 128);
#pragma unroll
        for (int t = 0; t < 2; ++t) s[t][j] = wmma_bf16(qf[t][ks], bk, s[t][j]);
      }

    // online softmax, write P (bf16) into per-wave LDS [32][64]
#pragma unroll
    for (int t = 0; t < 2; ++t) {
#pragma unroll
      for (int r = 0; r < 8; ++r) {
        float mx = fmaxf(fmaxf(s[t][0][r], s[t][1][r]), fmaxf(s[t][2][r], s[t][3][r]));
#pragma unroll
        for (int off = 1; off < 16; off <<= 1) mx = fmaxf(mx, __shfl_xor(mx, off));
        float mnew  = fmaxf(mrun[t][r], mx);
        float alpha = __expf(mrun[t][r] - mnew);
        mrun[t][r]  = mnew;
        const int rl = ((lane >> 4) << 3) + r;
        float rs = 0.0f;
#pragma unroll
        for (int j = 0; j < 4; ++j) {
          float p = __expf(s[t][j][r] - mnew);
          rs += p;
          lds_store2(PS + (unsigned)((t * 16 + rl) * 64 + j * 16 + (lane & 15)) * 2u,
                     f2bf(p));
        }
#pragma unroll
        for (int off = 1; off < 16; off <<= 1) rs += __shfl_xor(rs, off);
        lrun[t][r] = lrun[t][r] * alpha + rs;
#pragma unroll
        for (int nj = 0; nj < 8; ++nj) o[t][nj][r] *= alpha;
      }
    }

    // ctx += P @ V   (K=64 keys -> two K=32 WMMA steps)
#pragma unroll
    for (int ks2 = 0; ks2 < 2; ++ks2) {
      v16bf pa[2];
#pragma unroll
      for (int t = 0; t < 2; ++t) pa[t] = frag_a(PS, t * 16, ks2 * 32, 64);
#pragma unroll
      for (int nj = 0; nj < 8; ++nj) {
        v16bf vfrag = frag_b(VTS, nj * 16, ks2 * 32, 72);
#pragma unroll
        for (int t = 0; t < 2; ++t) o[t][nj] = wmma_bf16(pa[t], vfrag, o[t][nj]);
      }
    }

    if (it + 1 < 32) wait_async0();   // retire next tile's async copies
    __syncthreads();                  // publish them; fence VTs for next transpose
  }

  // normalize and write ctx bf16 as [B, L, D] (D index = h*128 + hd)
#pragma unroll
  for (int t = 0; t < 2; ++t)
#pragma unroll
    for (int nj = 0; nj < 8; ++nj)
#pragma unroll
      for (int r = 0; r < 8; ++r) {
        const int lrow = w * 32 + t * 16 + ((lane >> 4) << 3) + r;
        const int d    = h * 128 + nj * 16 + (lane & 15);
        float v = o[t][nj][r] / lrun[t][r];
        ctxo[(size_t)(b * 256 + lrow) * 2048 + d] = f2bf(v);
      }
}

// ---------- launcher ----------
extern "C" void kernel_launch(void* const* d_in, const int* in_sizes, int n_in,
                              void* d_out, int out_size, void* d_ws, size_t ws_size,
                              hipStream_t stream) {
  (void)in_sizes; (void)n_in; (void)out_size; (void)ws_size;
  const float* x   = (const float*)d_in[0];
  const float* enc = (const float*)d_in[1];
  const float* Wq  = (const float*)d_in[2];
  const float* bq  = (const float*)d_in[3];
  const float* Wk  = (const float*)d_in[4];
  const float* bk  = (const float*)d_in[5];
  const float* Wv  = (const float*)d_in[6];
  const float* bv  = (const float*)d_in[7];
  const float* Wo  = (const float*)d_in[8];
  const float* bo  = (const float*)d_in[9];

  char* ws = (char*)d_ws;
  const size_t QN = (size_t)4 * 16 * 256 * 128;    // 2Mi bf16
  const size_t KN = (size_t)4 * 16 * 2048 * 128;   // 16Mi bf16
  __bf16* Qb = reinterpret_cast<__bf16*>(ws);
  __bf16* Kb = reinterpret_cast<__bf16*>(ws + QN * 2);
  __bf16* Vb = reinterpret_cast<__bf16*>(ws + (QN + KN) * 2);
  __bf16* Cb = reinterpret_cast<__bf16*>(ws + (QN + 2 * KN) * 2);

  const float scale = 0.08838834764831845f;  // 1/sqrt(128)
  dim3 blk(256);
  const size_t gemm_lds = (size_t)(4 * 128 * 40) * 2;        // 40 KB
  const size_t attn_lds = (size_t)182272;                    // ~178 KB

  // Q projection + RoPE + scale  (M=1024, K=2048, N=2048)
  gemm_wmma<0, false><<<dim3(16, 8), blk, gemm_lds, stream>>>(
      x, nullptr, Wq, bq, Qb, 1024, 2048, 2048, 256, scale);
  // K projection  (M=8192, K=1024, N=2048)
  gemm_wmma<1, false><<<dim3(16, 64), blk, gemm_lds, stream>>>(
      enc, nullptr, Wk, bk, Kb, 8192, 1024, 2048, 2048, 1.0f);
  // V projection
  gemm_wmma<1, false><<<dim3(16, 64), blk, gemm_lds, stream>>>(
      enc, nullptr, Wv, bv, Vb, 8192, 1024, 2048, 2048, 1.0f);
  // fused flash attention, one block per (b,h)
  attn_fused<<<64, blk, attn_lds, stream>>>(Qb, Kb, Vb, Cb);
  // output projection (bf16 A, fp32 out)
  gemm_wmma<2, true><<<dim3(16, 8), blk, gemm_lds, stream>>>(
      nullptr, Cb, Wo, bo, d_out, 1024, 2048, 2048, 1, 1.0f);
}
